// GCN_alignment_joint_att_47502338294137
// MI455X (gfx1250) — matrix operand order repaired and checked by reading, not compiled, not run on hardware
//
#include <hip/hip_runtime.h>
#include <hip/hip_bf16.h>
#include <stdint.h>
#include <math.h>

// ---------------------------------------------------------------------------
// Problem constants
// ---------------------------------------------------------------------------
#define N_TOT 12000   // graph size
#define H_DIM 256     // hidden
#define B_SZ  128     // batch
#define L_TOK 64      // tokens per example
#define R_G   8       // gate rows
#define GH_W  350     // gate hidden width
#define GHP   352     // padded to multiple of 32 for WMMA K/N tiling

typedef __bf16 bf16_t;
typedef __attribute__((ext_vector_type(16))) __bf16 v16bf;
typedef __attribute__((ext_vector_type(8)))  __bf16 bf16x8;
typedef __attribute__((ext_vector_type(8)))  float  v8f;
typedef __attribute__((ext_vector_type(4)))  float  f32x4;

// ---------------------------------------------------------------------------
// CDNA5 helpers
// ---------------------------------------------------------------------------
__device__ __forceinline__ v8f wmma_bf16(v16bf a, v16bf b, v8f c) {
  // D = A(16x32 bf16) x B(32x16 bf16) + C(16x16 f32)
  return __builtin_amdgcn_wmma_f32_16x16x32_bf16(
      /*neg_a=*/false, a, /*neg_b=*/false, b,
      /*c_mod=*/(short)0, c, /*reuse_a=*/false, /*reuse_b=*/false);
}

__device__ __forceinline__ v8f zero8() {
  v8f z;
#pragma unroll
  for (int i = 0; i < 8; ++i) z[i] = 0.0f;
  return z;
}

// A-fragment (16x32, bf16), source row-major [M][K], ld in elements.
// Lane l (<16): row M=l, halves 0..7 = K0..7, 8..15 = K16..23.
// Lane l (>=16): row M=l-16, halves 0..7 = K8..15, 8..15 = K24..31.
__device__ __forceinline__ v16bf frag_a_rm(const bf16_t* __restrict__ p, int ld, int lane) {
  const bf16_t* base = p + (size_t)(lane & 15) * ld + ((lane >> 4) << 3);
  bf16x8 lo = *(const bf16x8*)(base);
  bf16x8 hi = *(const bf16x8*)(base + 16);
  v16bf f;
#pragma unroll
  for (int i = 0; i < 8; ++i) { f[i] = lo[i]; f[i + 8] = hi[i]; }
  return f;
}

// B-fragment (32x16, bf16), source N-major [N][K] (i.e. B^T row-major), ld in elems.
// Lane l: column N = l&15; halves 0..7 = K(8h..8h+7), 8..15 = K(16+8h..16+8h+7), h = l>>4.
__device__ __forceinline__ v16bf frag_b_nm(const bf16_t* __restrict__ p, int ld, int lane) {
  const bf16_t* base = p + (size_t)(lane & 15) * ld + ((lane >> 4) << 3);
  bf16x8 lo = *(const bf16x8*)(base);
  bf16x8 hi = *(const bf16x8*)(base + 16);
  v16bf f;
#pragma unroll
  for (int i = 0; i < 8; ++i) { f[i] = lo[i]; f[i + 8] = hi[i]; }
  return f;
}

// Async global -> LDS copy, 16 bytes per lane (CDNA5 GLOBAL_LOAD_ASYNC_TO_LDS_B128).
__device__ __forceinline__ void async_copy_b128(void* lds, const void* gptr) {
  uint32_t l = (uint32_t)(uintptr_t)lds;  // low 32 bits of flat shared addr == LDS offset
  asm volatile("global_load_async_to_lds_b128 %0, %1, off"
               :: "v"(l), "v"(gptr) : "memory");
}

template <int N>
__device__ __forceinline__ void wait_async() {
  asm volatile("s_wait_asynccnt %0" :: "i"(N) : "memory");
}

// ---------------------------------------------------------------------------
// K0a: embedding [N][H] f32  ->  ET [H][N] bf16  (LDS 32x32 tile transpose)
// ---------------------------------------------------------------------------
__global__ __launch_bounds__(256) void k_emb_t(const float* __restrict__ emb,
                                               bf16_t* __restrict__ ET) {
  __shared__ float tile[32][33];
  const int n0 = blockIdx.x * 32;
  const int h0 = blockIdx.y * 32;
  const int tx = threadIdx.x & 31;
  const int ty = threadIdx.x >> 5;  // 0..7
#pragma unroll
  for (int i = 0; i < 4; ++i) {
    int r = ty + i * 8;
    tile[r][tx] = emb[(size_t)(n0 + r) * H_DIM + h0 + tx];
  }
  __syncthreads();
#pragma unroll
  for (int i = 0; i < 4; ++i) {
    int r = ty + i * 8;  // h-row within tile
    ET[(size_t)(h0 + r) * N_TOT + n0 + tx] = (bf16_t)tile[tx][r];
  }
}

// ---------------------------------------------------------------------------
// K0b: W1 [H][GH] f32 -> W1T [GHP][H] bf16 (rows >= GH zero-padded)
// ---------------------------------------------------------------------------
__global__ __launch_bounds__(256) void k_w1_t(const float* __restrict__ W1,
                                              bf16_t* __restrict__ W1T) {
  const int g = blockIdx.x;    // 0..351
  const int h = threadIdx.x;   // 0..255
  float v = (g < GH_W) ? W1[(size_t)h * GH_W + g] : 0.0f;
  W1T[(size_t)g * H_DIM + h] = (bf16_t)v;
}

// ---------------------------------------------------------------------------
// K1: layer1 = A @ embedding   (12000x12000 x 12000x256), bf16 WMMA
// Block = 256 threads (8 waves), tile BM=128 x BN=128, BK=32, double-buffered
// LDS filled via async global->LDS (A tile raw f32, B tile bf16 from ET).
// Grid: (2 col-blocks, 94 row-blocks); x-fastest order keeps the two column
// blocks of a row co-resident so A streams from HBM once (2nd read hits L2).
// ---------------------------------------------------------------------------
#define BM 128
#define BN 128
#define BK 32
#define AS_LD 36   // f32 row stride (144B: 16B aligned, conflict-free frags)
#define BS_LD 40   // bf16 row stride (80B: 16B aligned, conflict-free frags)

__global__ __launch_bounds__(256) void k_gemm1(const float* __restrict__ A,
                                               const bf16_t* __restrict__ ET,
                                               float* __restrict__ L1) {
  __shared__ __attribute__((aligned(16))) float  As[2][BM][AS_LD];
  __shared__ __attribute__((aligned(16))) bf16_t Bs[2][BN][BS_LD];

  const int t = threadIdx.x;
  const int lane = t & 31;
  const int wave = t >> 5;
  const int wr = wave >> 2;   // 0..1 (row half: 64 rows each)
  const int wc = wave & 3;    // 0..3 (col quarter: 32 cols each)
  const int row0 = blockIdx.y * BM;
  const int col0 = blockIdx.x * BN;

  v8f acc[4][2];
#pragma unroll
  for (int mt = 0; mt < 4; ++mt)
#pragma unroll
    for (int nt = 0; nt < 2; ++nt) acc[mt][nt] = zero8();

  auto issue = [&](int buf, int k0) {
    // A tile: 128 rows x 32 f32 = 1024 x 16B ops -> 4 per thread
#pragma unroll
    for (int i = 0; i < 4; ++i) {
      int idx = t + 256 * i;
      int m = idx >> 3, q = idx & 7;
      int gm = row0 + m; if (gm > N_TOT - 1) gm = N_TOT - 1;
      async_copy_b128(&As[buf][m][q * 4],
                      A + (size_t)gm * N_TOT + k0 + q * 4);
    }
    // B tile from ET[h][k]: 128 h-rows x 32 bf16 = 512 x 16B ops -> 2/thread
#pragma unroll
    for (int i = 0; i < 2; ++i) {
      int idx = t + 256 * i;
      int h = idx >> 2, q = idx & 3;
      async_copy_b128(&Bs[buf][h][q * 8],
                      ET + (size_t)(col0 + h) * N_TOT + k0 + q * 8);
    }
  };

  const int KSTEPS = N_TOT / BK;  // 375
  issue(0, 0);
  issue(1, BK);

  for (int ks = 0; ks < KSTEPS; ++ks) {
    const int cur = ks & 1;
    if (ks + 1 < KSTEPS) wait_async<6>(); else wait_async<0>();
    __syncthreads();

    // A fragments: read f32 from LDS, convert to bf16 register layout
    v16bf afr[4];
#pragma unroll
    for (int mt = 0; mt < 4; ++mt) {
      const float* pa = &As[cur][wr * 64 + mt * 16 + (lane & 15)][(lane >> 4) * 8];
      f32x4 x0 = *(const f32x4*)(pa);
      f32x4 x1 = *(const f32x4*)(pa + 4);
      f32x4 x2 = *(const f32x4*)(pa + 16);
      f32x4 x3 = *(const f32x4*)(pa + 20);
      v16bf f;
#pragma unroll
      for (int i = 0; i < 4; ++i) {
        f[i]      = (bf16_t)x0[i];
        f[i + 4]  = (bf16_t)x1[i];
        f[i + 8]  = (bf16_t)x2[i];
        f[i + 12] = (bf16_t)x3[i];
      }
      afr[mt] = f;
    }

#pragma unroll
    for (int nt = 0; nt < 2; ++nt) {
      const bf16_t* pb = &Bs[cur][wc * 32 + nt * 16 + (lane & 15)][(lane >> 4) * 8];
      bf16x8 b0 = *(const bf16x8*)(pb);
      bf16x8 b1 = *(const bf16x8*)(pb + 16);
      v16bf bfr;
#pragma unroll
      for (int i = 0; i < 8; ++i) { bfr[i] = b0[i]; bfr[i + 8] = b1[i]; }
#pragma unroll
      for (int mt = 0; mt < 4; ++mt)
        acc[mt][nt] = wmma_bf16(afr[mt], bfr, acc[mt][nt]);
    }

    __syncthreads();
    if (ks + 2 < KSTEPS) issue(cur, (ks + 2) * BK);
  }

  // Epilogue: C layout -> VGPR i holds M=i (lanes 0-15) / M=i+8 (lanes 16-31)
#pragma unroll
  for (int mt = 0; mt < 4; ++mt)
#pragma unroll
    for (int nt = 0; nt < 2; ++nt)
#pragma unroll
      for (int i = 0; i < 8; ++i) {
        int m = row0 + wr * 64 + mt * 16 + i + ((lane >> 4) << 3);
        int n = col0 + wc * 32 + nt * 16 + (lane & 15);
        if (m < N_TOT) L1[(size_t)m * H_DIM + n] = acc[mt][nt][i];
      }
}

// ---------------------------------------------------------------------------
// K2: gather rows -> Xf (f32 [B][L][H]), Xb (bf16 [B][L][H]), XT (bf16 [B][H][L])
// ---------------------------------------------------------------------------
__global__ __launch_bounds__(256) void k_gather(const float* __restrict__ L1,
                                                const int* __restrict__ idx,
                                                float* __restrict__ Xf,
                                                bf16_t* __restrict__ Xb,
                                                bf16_t* __restrict__ XT) {
  const int b = blockIdx.x;
  const int h = threadIdx.x;
  for (int l = 0; l < L_TOK; ++l) {
    int row = idx[b * L_TOK + l];
    float v = L1[(size_t)row * H_DIM + h];
    size_t o = ((size_t)b * L_TOK + l) * H_DIM + h;
    Xf[o] = v;
    bf16_t bv = (bf16_t)v;
    Xb[o] = bv;
    XT[(size_t)b * H_DIM * L_TOK + (size_t)h * L_TOK + l] = bv;
  }
}

// ---------------------------------------------------------------------------
// K3: score = Q @ Av^T (64x64, K=256) via WMMA, then dual softmax:
//   wq [b][l][m] = softmax over m;  wa [b][m][l] = softmax over l. (bf16 out)
// Block: 128 threads (4 waves); wave w owns n-tile w.
// ---------------------------------------------------------------------------
__global__ __launch_bounds__(128) void k_score(const bf16_t* __restrict__ Qb,
                                               const bf16_t* __restrict__ Avb,
                                               bf16_t* __restrict__ wq,
                                               bf16_t* __restrict__ wa) {
  __shared__ float sc[64][65];
  const int b = blockIdx.x;
  const int t = threadIdx.x, lane = t & 31, w = t >> 5;
  const bf16_t* Qp = Qb  + (size_t)b * L_TOK * H_DIM;
  const bf16_t* Ap = Avb + (size_t)b * L_TOK * H_DIM;

  v8f acc[4];
#pragma unroll
  for (int mt = 0; mt < 4; ++mt) acc[mt] = zero8();

  for (int k0 = 0; k0 < H_DIM; k0 += 32) {
    v16bf bfr = frag_b_nm(Ap + (size_t)(w * 16) * H_DIM + k0, H_DIM, lane);
#pragma unroll
    for (int mt = 0; mt < 4; ++mt) {
      v16bf af = frag_a_rm(Qp + (size_t)(mt * 16) * H_DIM + k0, H_DIM, lane);
      acc[mt] = wmma_bf16(af, bfr, acc[mt]);
    }
  }
#pragma unroll
  for (int mt = 0; mt < 4; ++mt)
#pragma unroll
    for (int i = 0; i < 8; ++i)
      sc[mt * 16 + i + ((lane >> 4) << 3)][w * 16 + (lane & 15)] = acc[mt][i];
  __syncthreads();

  if (t < 64) {
    // wq: softmax over m for row l = t
    {
      int l = t;
      float mx = -1e30f;
      for (int m = 0; m < 64; ++m) mx = fmaxf(mx, sc[l][m]);
      float s = 0.f;
      for (int m = 0; m < 64; ++m) s += expf(sc[l][m] - mx);
      float inv = 1.f / s;
      for (int m = 0; m < 64; ++m)
        wq[((size_t)b * 64 + l) * 64 + m] = (bf16_t)(expf(sc[l][m] - mx) * inv);
    }
    // wa: softmax over l for column m = t
    {
      int m = t;
      float mx = -1e30f;
      for (int l = 0; l < 64; ++l) mx = fmaxf(mx, sc[l][m]);
      float s = 0.f;
      for (int l = 0; l < 64; ++l) s += expf(sc[l][m] - mx);
      float inv = 1.f / s;
      for (int l = 0; l < 64; ++l)
        wa[((size_t)b * 64 + m) * 64 + l] = (bf16_t)(expf(sc[l][m] - mx) * inv);
    }
  }
}

// ---------------------------------------------------------------------------
// K4: Out[64][256] = Wm[64][64] @ X  where B-operand given N-major as XT[256][64].
// Used for EQ = wq @ Av (XT = AvT) and EA = wa @ Q (XT = QbT).
// Block: 128 threads (4 waves); wave w owns n-tiles {w, w+4, w+8, w+12}.
// ---------------------------------------------------------------------------
__global__ __launch_bounds__(128) void k_attmix(const bf16_t* __restrict__ Wm,
                                                const bf16_t* __restrict__ XT,
                                                float* __restrict__ Out) {
  const int b = blockIdx.x;
  const int t = threadIdx.x, lane = t & 31, w = t >> 5;
  const bf16_t* Wp = Wm + (size_t)b * 64 * 64;
  const bf16_t* Tp = XT + (size_t)b * H_DIM * 64;

  v8f acc[4][4];
#pragma unroll
  for (int mt = 0; mt < 4; ++mt)
#pragma unroll
    for (int j = 0; j < 4; ++j) acc[mt][j] = zero8();

#pragma unroll
  for (int k0 = 0; k0 < 64; k0 += 32) {
    v16bf af[4];
#pragma unroll
    for (int mt = 0; mt < 4; ++mt)
      af[mt] = frag_a_rm(Wp + (size_t)(mt * 16) * 64 + k0, 64, lane);
#pragma unroll
    for (int j = 0; j < 4; ++j) {
      int ntile = w + 4 * j;
      v16bf bfr = frag_b_nm(Tp + (size_t)(ntile * 16) * 64 + k0, 64, lane);
#pragma unroll
      for (int mt = 0; mt < 4; ++mt)
        acc[mt][j] = wmma_bf16(af[mt], bfr, acc[mt][j]);
    }
  }
#pragma unroll
  for (int mt = 0; mt < 4; ++mt)
#pragma unroll
    for (int j = 0; j < 4; ++j)
#pragma unroll
      for (int i = 0; i < 8; ++i) {
        int m = mt * 16 + i + ((lane >> 4) << 3);
        int n = (w + 4 * j) * 16 + (lane & 15);
        Out[((size_t)b * 64 + m) * H_DIM + n] = acc[mt][j][i];
      }
}

// ---------------------------------------------------------------------------
// K5: gate(X) = softmax_l( tanh(X @ W1) @ W2 )  -> G f32 [B][L][R]
// Phase 1: hidden = tanh(Xb @ W1) via WMMA (K=256, N tiles over GHP=352) -> LDS
// Phase 2: hidden @ W2 (K=350, N=8) via VALU
// Phase 3: softmax over token dim l, per column r
// ---------------------------------------------------------------------------
__global__ __launch_bounds__(128) void k_gate(const bf16_t* __restrict__ Xb,
                                              const bf16_t* __restrict__ W1T,
                                              const float* __restrict__ W2,
                                              float* __restrict__ G) {
  __shared__ bf16_t h1[64][GHP + 8];   // 64 x 360 bf16
  __shared__ float  w2s[GH_W][R_G];
  __shared__ float  gp[64][9];
  const int b = blockIdx.x;
  const int t = threadIdx.x, lane = t & 31, w = t >> 5;

  for (int i = t; i < GH_W * R_G; i += 128) ((float*)w2s)[i] = W2[i];

  const bf16_t* Xp = Xb + (size_t)b * L_TOK * H_DIM;

  // Phase 1
  for (int j = 0; j < 6; ++j) {
    int ntile = w + 4 * j;          // wave-uniform
    if (ntile < GHP / 16) {
      v8f acc[4];
#pragma unroll
      for (int mt = 0; mt < 4; ++mt) acc[mt] = zero8();
      for (int k0 = 0; k0 < H_DIM; k0 += 32) {
        v16bf bfr = frag_b_nm(W1T + (size_t)(ntile * 16) * H_DIM + k0, H_DIM, lane);
#pragma unroll
        for (int mt = 0; mt < 4; ++mt) {
          v16bf af = frag_a_rm(Xp + (size_t)(mt * 16) * H_DIM + k0, H_DIM, lane);
          acc[mt] = wmma_bf16(af, bfr, acc[mt]);
        }
      }
#pragma unroll
      for (int mt = 0; mt < 4; ++mt)
#pragma unroll
        for (int i = 0; i < 8; ++i) {
          int m = mt * 16 + i + ((lane >> 4) << 3);
          int g = ntile * 16 + (lane & 15);
          h1[m][g] = (bf16_t)tanhf(acc[mt][i]);
        }
    }
  }
  __syncthreads();

  // Phase 2: g_pre[l][r] = sum_g h1[l][g] * W2[g][r]
  {
    const int l = t & 63;
    const int rg = t >> 6;  // 0..1 -> r in [rg*4, rg*4+4)
    float accp[4] = {0.f, 0.f, 0.f, 0.f};
    for (int g = 0; g < GH_W; ++g) {
      float hv = (float)h1[l][g];
#pragma unroll
      for (int rr = 0; rr < 4; ++rr) accp[rr] += hv * w2s[g][rg * 4 + rr];
    }
#pragma unroll
    for (int rr = 0; rr < 4; ++rr) gp[l][rg * 4 + rr] = accp[rr];
  }
  __syncthreads();

  // Phase 3: softmax over l, per r
  if (t < R_G) {
    int r = t;
    float mx = -1e30f;
    for (int l = 0; l < 64; ++l) mx = fmaxf(mx, gp[l][r]);
    float s = 0.f;
    for (int l = 0; l < 64; ++l) s += expf(gp[l][r] - mx);
    float inv = 1.f / s;
    for (int l = 0; l < 64; ++l)
      G[((size_t)b * 64 + l) * R_G + r] = expf(gp[l][r] - mx) * inv;
  }
}

// ---------------------------------------------------------------------------
// K6: tq/ta contraction + final linear + log_softmax -> out [B][2]
// ---------------------------------------------------------------------------
__global__ __launch_bounds__(256) void k_final(const float* __restrict__ Qf,
                                               const float* __restrict__ EQ,
                                               const float* __restrict__ Gq,
                                               const float* __restrict__ Avf,
                                               const float* __restrict__ EA,
                                               const float* __restrict__ Ga,
                                               const float* __restrict__ linW,
                                               const float* __restrict__ linb,
                                               float* __restrict__ out) {
  __shared__ float gq_s[64][9], ga_s[64][9];
  __shared__ float red[256];
  __shared__ float scs[2];
  const int b = blockIdx.x, t = threadIdx.x;

  for (int i = t; i < 64 * R_G; i += 256) {
    int l = i >> 3, r = i & 7;
    gq_s[l][r] = Gq[((size_t)b * 64 + l) * R_G + r];
    ga_s[l][r] = Ga[((size_t)b * 64 + l) * R_G + r];
  }
  __syncthreads();

  const int h = t;  // 0..255
  float tq[8], ta[8];
#pragma unroll
  for (int r = 0; r < 8; ++r) { tq[r] = 0.f; ta[r] = 0.f; }

  for (int l = 0; l < L_TOK; ++l) {
    size_t off = ((size_t)b * L_TOK + l) * H_DIM + h;
    float dq = Qf[off] - EQ[off];  dq *= dq;
    float da = Avf[off] - EA[off]; da *= da;
#pragma unroll
    for (int r = 0; r < 8; ++r) {
      tq[r] += gq_s[l][r] * dq;
      ta[r] += ga_s[l][r] * da;
    }
  }

  // re[b][r*512 + h] = tq[r][h];  re[b][r*512 + 256 + h] = ta[r][h]
  float p0 = 0.f, p1 = 0.f;
#pragma unroll
  for (int r = 0; r < 8; ++r) {
    const float* wq_ = linW + ((size_t)(r * 512) + h) * 2;
    p0 += tq[r] * wq_[0];
    p1 += tq[r] * wq_[1];
    const float* wa_ = linW + ((size_t)(r * 512) + 256 + h) * 2;
    p0 += ta[r] * wa_[0];
    p1 += ta[r] * wa_[1];
  }

  red[t] = p0; __syncthreads();
  for (int s = 128; s > 0; s >>= 1) { if (t < s) red[t] += red[t + s]; __syncthreads(); }
  if (t == 0) scs[0] = red[0] + linb[0];
  __syncthreads();
  red[t] = p1; __syncthreads();
  for (int s = 128; s > 0; s >>= 1) { if (t < s) red[t] += red[t + s]; __syncthreads(); }
  if (t == 0) {
    float a0 = scs[0], a1 = red[0] + linb[1];
    float m = fmaxf(a0, a1);
    float lse = m + logf(expf(a0 - m) + expf(a1 - m));
    out[b * 2 + 0] = a0 - lse;
    out[b * 2 + 1] = a1 - lse;
  }
}

// ---------------------------------------------------------------------------
// Launch
// ---------------------------------------------------------------------------
extern "C" void kernel_launch(void* const* d_in, const int* in_sizes, int n_in,
                              void* d_out, int out_size, void* d_ws, size_t ws_size,
                              hipStream_t stream) {
  const float* A    = (const float*)d_in[0];
  const float* emb  = (const float*)d_in[1];
  const float* W1   = (const float*)d_in[2];
  const float* W2   = (const float*)d_in[3];
  const float* linW = (const float*)d_in[4];
  const float* linb = (const float*)d_in[5];
  const int* idx_q  = (const int*)d_in[6];
  const int* idx_a  = (const int*)d_in[7];
  float* out = (float*)d_out;

  char* ws = (char*)d_ws;
  size_t off = 0;
  auto alloc = [&](size_t bytes) -> char* {
    char* p = ws + off;
    off += (bytes + 255) & ~(size_t)255;
    return p;
  };

  bf16_t* ET   = (bf16_t*)alloc((size_t)H_DIM * N_TOT * sizeof(bf16_t));   // 6.1 MB
  float*  L1   = (float*) alloc((size_t)N_TOT * H_DIM * sizeof(float));    // 12.3 MB
  bf16_t* W1T  = (bf16_t*)alloc((size_t)GHP * H_DIM * sizeof(bf16_t));
  float*  Qf   = (float*) alloc((size_t)B_SZ * L_TOK * H_DIM * sizeof(float));
  float*  Avf  = (float*) alloc((size_t)B_SZ * L_TOK * H_DIM * sizeof(float));
  bf16_t* Qb   = (bf16_t*)alloc((size_t)B_SZ * L_TOK * H_DIM * sizeof(bf16_t));
  bf16_t* Avb  = (bf16_t*)alloc((size_t)B_SZ * L_TOK * H_DIM * sizeof(bf16_t));
  bf16_t* QbT  = (bf16_t*)alloc((size_t)B_SZ * H_DIM * L_TOK * sizeof(bf16_t));
  bf16_t* AvT  = (bf16_t*)alloc((size_t)B_SZ * H_DIM * L_TOK * sizeof(bf16_t));
  bf16_t* wqb  = (bf16_t*)alloc((size_t)B_SZ * 64 * 64 * sizeof(bf16_t));
  bf16_t* wab  = (bf16_t*)alloc((size_t)B_SZ * 64 * 64 * sizeof(bf16_t));
  float*  EQ   = (float*) alloc((size_t)B_SZ * L_TOK * H_DIM * sizeof(float));
  float*  EA   = (float*) alloc((size_t)B_SZ * L_TOK * H_DIM * sizeof(float));
  float*  Gq   = (float*) alloc((size_t)B_SZ * L_TOK * R_G * sizeof(float));
  float*  Ga   = (float*) alloc((size_t)B_SZ * L_TOK * R_G * sizeof(float));
  (void)ws_size; (void)in_sizes; (void)n_in; (void)out_size;

  k_emb_t<<<dim3(N_TOT / 32, H_DIM / 32), 256, 0, stream>>>(emb, ET);
  k_w1_t<<<GHP, H_DIM, 0, stream>>>(W1, W1T);
  k_gemm1<<<dim3(H_DIM / BN, (N_TOT + BM - 1) / BM), 256, 0, stream>>>(A, ET, L1);
  k_gather<<<B_SZ, H_DIM, 0, stream>>>(L1, idx_q, Qf, Qb, QbT);
  k_gather<<<B_SZ, H_DIM, 0, stream>>>(L1, idx_a, Avf, Avb, AvT);
  k_score<<<B_SZ, 128, 0, stream>>>(Qb, Avb, wqb, wab);
  k_attmix<<<B_SZ, 128, 0, stream>>>(wqb, AvT, EQ);
  k_attmix<<<B_SZ, 128, 0, stream>>>(wab, QbT, EA);
  k_gate<<<B_SZ, 128, 0, stream>>>(Qb, W1T, W2, Gq);
  k_gate<<<B_SZ, 128, 0, stream>>>(Avb, W1T, W2, Ga);
  k_final<<<B_SZ, 256, 0, stream>>>(Qf, EQ, Gq, Avf, EA, Ga, linW, linb, out);
}
